// ContextualAttentionModule_9337258902269
// MI455X (gfx1250) — matrix operand order repaired and checked by reading, not compiled, not run on hardware
//
#include <hip/hip_runtime.h>
#include <hip/hip_bf16.h>
#include <math.h>

// ---------------------------------------------------------------------------
// Contextual attention for MI455X (gfx1250, wave32, WMMA).
// Both large matmuls (correlation 2304x864x2304 and deconv-as-GEMM
// 2304x2304x1536, per batch of 4) run on v_wmma_f32_16x16x32_f16 with f16
// fragments pre-packed in the exact CDNA5 lane layout.  The GEMM waves are
// register-blocked 2x4 (8 WMMAs per 6 fragment loads) so the kernel is not
// L2-load-bound: 1.5 b128 loads per v_wmma instead of 4.
// ---------------------------------------------------------------------------

typedef __attribute__((ext_vector_type(16))) _Float16 v16h;
typedef __attribute__((ext_vector_type(8)))  float    v8f;

#define NB 4
#define CC 96
#define HH 96
#define WW 96
#define HS 48                    // downsampled spatial (::2)
#define LL 2304                  // 48*48
#define KDIM1 864                // C * 3*3
#define KT1 27                   // 864/32
#define MT1 144                  // 2304/16
#define NT1 144
#define KT2 72                   // 2304/32
#define MT2 144
#define NT2 96                   // 1536/16
#define CF  1536                 // C * 4*4
#define OUT0 ((size_t)NB*CC*HH*WW)   // 3,538,944 floats, then att_score

// workspace layout (bytes, all 256-aligned); buffers re-used across phases
#define NORM_OFF  ((size_t)0)
#define BIAS_OFF  ((size_t)36864)
#define A1_OFF    ((size_t)73728)                       // 15,925,248 B
#define B1_OFF    (A1_OFF + (size_t)15925248)           // 15,925,248 B
#define X_OFF     (B1_OFF + (size_t)15925248)           // 84,934,656 B (corr -> score -> V)
#define Y_OFF     (X_OFF + (size_t)84934656)            // 84,934,656 B (m3 -> A2 frags)
#define WS_NEEDED (Y_OFF + (size_t)84934656)            // ~192.4 MiB
#define A2_OFF    Y_OFF
#define B2_OFF    A1_OFF
#define V_OFF     X_OFF
#define SCORE_OFF X_OFF

// ---------------- 2x4 register-blocked WMMA GEMM:  C = A * B ----------------
// A frags: [n][mt][kt][lane][16 f16]   B frags: [n][kt][nt][lane][16 f16]
// Each wave owns a 32x64 block of C (2x4 tiles of 16x16).
__global__ __launch_bounds__(256)
void wmma_gemm_tile(const v16h* __restrict__ A, const v16h* __restrict__ B,
                    float* __restrict__ Cm, int MT, int NT, int KT, int ldc,
                    long long cStride, int tilesPerBatch, int totalWaves) {
  int wave = blockIdx.x * (blockDim.x >> 5) + (threadIdx.x >> 5);
  if (wave >= totalWaves) return;          // wave-uniform: EXEC stays all-ones
  int lane = threadIdx.x & 31;
  int NTb = NT >> 2;                       // N tile-blocks of 4
  int n   = wave / tilesPerBatch;
  int t   = wave - n * tilesPerBatch;
  int mt  = (t / NTb) * 2;                 // 2 M tiles per wave
  int nt  = (t % NTb) * 4;                 // 4 N tiles per wave

  long long aBase = ((long long)n * MT * KT + (long long)mt * KT) * 32 + lane;
  long long bBase = ((long long)n * KT * NT + nt) * 32 + lane;
  long long aStep = (long long)KT * 32;    // stride between consecutive mt
  v8f acc[2][4] = {};
  for (int kt = 0; kt < KT; ++kt) {
    v16h a[2], b[4];
#pragma unroll
    for (int i = 0; i < 2; ++i)
      a[i] = A[aBase + (long long)kt * 32 + i * aStep];
#pragma unroll
    for (int j = 0; j < 4; ++j)
      b[j] = B[bBase + (long long)kt * NT * 32 + j * 32];
#pragma unroll
    for (int i = 0; i < 2; ++i)
#pragma unroll
      for (int j = 0; j < 4; ++j)
        acc[i][j] = __builtin_amdgcn_wmma_f32_16x16x32_f16(
            false, a[i], false, b[j], (short)0, acc[i][j], false, false);
  }
  float* cp = Cm + (long long)n * cStride;
#pragma unroll
  for (int i = 0; i < 2; ++i) {
    int rowBase = (mt + i) * 16 + ((lane >> 4) << 3);
#pragma unroll
    for (int j = 0; j < 4; ++j) {
      int col = (nt + j) * 16 + (lane & 15);
#pragma unroll
      for (int v = 0; v < 8; ++v)
        cp[(long long)(rowBase + v) * ldc + col] = acc[i][j][v];
    }
  }
}

// ---------------- phase 0: patch norms + mask bias --------------------------
__global__ __launch_bounds__(256)
void k_norm_bias(const float* __restrict__ ctx, const float* __restrict__ mask,
                 float* __restrict__ norm, float* __restrict__ bias) {
  int tid = blockIdx.x * blockDim.x + threadIdx.x;
  if (tid >= NB * LL) return;
  int l = tid % LL, n = tid / LL;
  int u = l / HS, v = l % HS;
  float s = 0.f;
  for (int kk = 0; kk < KDIM1; ++kk) {
    int c = kk / 9, r = kk % 9;
    int ui = u + r / 3 - 1, vj = v + r % 3 - 1;
    if (ui >= 0 && ui < HS && vj >= 0 && vj < HS) {
      float val = ctx[(((size_t)n * CC + c) * HH + 2 * ui) * WW + 2 * vj];
      s += val * val;
    }
  }
  float msum = 0.f;
  for (int r = 0; r < 9; ++r) {
    int ui = u + r / 3 - 1, vj = v + r % 3 - 1;
    if (ui >= 0 && ui < HS && vj >= 0 && vj < HS)
      msum += mask[((size_t)n * HH + 2 * ui) * WW + 2 * vj];
  }
  norm[tid] = sqrtf(s);
  bias[tid] = (msum > 0.f) ? -1.0e9f : 0.f;
}

// CDNA5 f16 A-fragment K map (16x32): K = half*8 + j + (j>=8 ? 8 : 0)
__device__ __forceinline__ int amapK(int half, int j) {
  return half * 8 + j + ((j >= 8) ? 8 : 0);
}

// ---------------- phase 1: build GEMM1 fragments ----------------------------
__global__ __launch_bounds__(256)
void k_build_A1(const float* __restrict__ ctx, const float* __restrict__ norm,
                v16h* __restrict__ out) {
  int tid = blockIdx.x * blockDim.x + threadIdx.x;
  if (tid >= NB * MT1 * KT1 * 32) return;
  int lane = tid & 31, rest = tid >> 5;
  int kt = rest % KT1; rest /= KT1;
  int mt = rest % MT1; int n = rest / MT1;
  int l = mt * 16 + (lane & 15), half = lane >> 4;
  int u = l / HS, v = l % HS;
  float inv = 1.f / fmaxf(norm[n * LL + l], 1e-4f);
  v16h o;
#pragma unroll
  for (int j = 0; j < 16; ++j) {
    int K = kt * 32 + amapK(half, j);
    int c = K / 9, r = K % 9;
    int ui = u + r / 3 - 1, vj = v + r % 3 - 1;
    float val = 0.f;
    if (ui >= 0 && ui < HS && vj >= 0 && vj < HS)
      val = ctx[(((size_t)n * CC + c) * HH + 2 * ui) * WW + 2 * vj] * inv;
    o[j] = (_Float16)val;
  }
  out[tid] = o;
}

__global__ __launch_bounds__(256)
void k_build_B1(const float* __restrict__ x, v16h* __restrict__ out) {
  int tid = blockIdx.x * blockDim.x + threadIdx.x;
  if (tid >= NB * KT1 * NT1 * 32) return;
  int lane = tid & 31, rest = tid >> 5;
  int nt = rest % NT1; rest /= NT1;
  int kt = rest % KT1; int n = rest / KT1;
  int p = nt * 16 + (lane & 15), half = lane >> 4;
  int h = p / HS, w = p % HS;
  v16h o;
#pragma unroll
  for (int j = 0; j < 16; ++j) {
    int K = kt * 32 + amapK(half, j);        // B-fragment K map (32x16)
    int c = K / 9, r = K % 9;
    int hi = h + r / 3 - 1, wj = w + r % 3 - 1;
    float val = 0.f;
    if (hi >= 0 && hi < HS && wj >= 0 && wj < HS)
      val = x[(((size_t)n * CC + c) * HH + 2 * hi) * WW + 2 * wj];
    o[j] = (_Float16)val;
  }
  out[tid] = o;
}

// ---------------- phase 3: fuse pass 1 (with interleaved transpose) ---------
// m1[P,Q] = corr[Q,P];  m3[P',Q'] = sum_d m1[P+d,Q+d], P=(P'%48)*48+P'/48 etc.
__global__ __launch_bounds__(256)
void k_fuse1(const float* __restrict__ corr0, float* __restrict__ m3) {
  long long tid = (long long)blockIdx.x * blockDim.x + threadIdx.x;
  if (tid >= (long long)NB * LL * LL) return;
  int Qp = (int)(tid % LL);
  int Pp = (int)((tid / LL) % LL);
  int n  = (int)(tid / ((long long)LL * LL));
  int P = (Pp % HS) * HS + Pp / HS;
  int Q = (Qp % HS) * HS + Qp / HS;
  float s = 0.f;
#pragma unroll
  for (int d = -1; d <= 1; ++d) {
    int Pd = P + d, Qd = Q + d;
    if (Pd >= 0 && Pd < LL && Qd >= 0 && Qd < LL)
      s += corr0[((size_t)n * LL + Qd) * LL + Pd];
  }
  m3[tid] = s;
}

// ---------------- phase 4: fuse pass 2 + un-transpose + mask bias -----------
// score[l,p] = bias[l] + sum_d m3[(w*48+h)+d, (v*48+u)+d]
__global__ __launch_bounds__(256)
void k_fuse2(const float* __restrict__ m3, const float* __restrict__ bias,
             float* __restrict__ score) {
  long long tid = (long long)blockIdx.x * blockDim.x + threadIdx.x;
  if (tid >= (long long)NB * LL * LL) return;
  int p = (int)(tid % LL);
  int l = (int)((tid / LL) % LL);
  int n = (int)(tid / ((long long)LL * LL));
  int u = l / HS, v = l % HS, h = p / HS, w = p % HS;
  int Pp = w * HS + h, Qp = v * HS + u;
  float s = bias[n * LL + l];
#pragma unroll
  for (int d = -1; d <= 1; ++d) {
    int Pd = Pp + d, Qd = Qp + d;
    if (Pd >= 0 && Pd < LL && Qd >= 0 && Qd < LL)
      s += m3[((size_t)n * LL + Pd) * LL + Qd];
  }
  score[tid] = s;
}

// ---------------- phase 5: softmax over l, writes att_score output ----------
__global__ __launch_bounds__(256)
void k_softmax(const float* __restrict__ score, float* __restrict__ att) {
  __shared__ float red[256];
  int n = blockIdx.x / LL, p = blockIdx.x % LL;
  const float* sc = score + (size_t)n * LL * LL + p;
  float m = -3.4e38f;
  for (int l = threadIdx.x; l < LL; l += blockDim.x)
    m = fmaxf(m, sc[(size_t)l * LL]);
  red[threadIdx.x] = m; __syncthreads();
  for (int o = 128; o > 0; o >>= 1) {
    if (threadIdx.x < (unsigned)o)
      red[threadIdx.x] = fmaxf(red[threadIdx.x], red[threadIdx.x + o]);
    __syncthreads();
  }
  m = red[0]; __syncthreads();
  float ssum = 0.f;
  for (int l = threadIdx.x; l < LL; l += blockDim.x)
    ssum += __expf(10.f * sc[(size_t)l * LL] - 10.f * m);
  red[threadIdx.x] = ssum; __syncthreads();
  for (int o = 128; o > 0; o >>= 1) {
    if (threadIdx.x < (unsigned)o)
      red[threadIdx.x] += red[threadIdx.x + o];
    __syncthreads();
  }
  float rinv = 1.f / red[0];
  float* ao = att + (size_t)n * LL * LL + p;
  for (int l = threadIdx.x; l < LL; l += blockDim.x)
    ao[(size_t)l * LL] = __expf(10.f * sc[(size_t)l * LL] - 10.f * m) * rinv;
}

// ---------------- phase 6: build GEMM2 fragments ----------------------------
// A2 = att^T : M=p (2304), K=l (2304)
__global__ __launch_bounds__(256)
void k_build_A2(const float* __restrict__ att, v16h* __restrict__ out) {
  int tid = blockIdx.x * blockDim.x + threadIdx.x;
  if (tid >= NB * MT2 * KT2 * 32) return;
  int lane = tid & 31, rest = tid >> 5;
  int kt = rest % KT2; rest /= KT2;
  int mt = rest % MT2; int n = rest / MT2;
  int p = mt * 16 + (lane & 15), half = lane >> 4;
  v16h o;
#pragma unroll
  for (int j = 0; j < 16; ++j) {
    int Kl = kt * 32 + amapK(half, j);
    o[j] = (_Float16)att[((size_t)n * LL + Kl) * LL + p];
  }
  out[tid] = o;
}

// B2 = raw_filter : K=l (2304), N=(c,ki,kj) (1536); rf value = ctx patch
__global__ __launch_bounds__(256)
void k_build_B2(const float* __restrict__ ctx, v16h* __restrict__ out) {
  int tid = blockIdx.x * blockDim.x + threadIdx.x;
  if (tid >= NB * KT2 * NT2 * 32) return;
  int lane = tid & 31, rest = tid >> 5;
  int nt = rest % NT2; rest /= NT2;
  int kt = rest % KT2; int n = rest / KT2;
  int cf = nt * 16 + (lane & 15);
  int c = cf >> 4, rem = cf & 15, ki = rem >> 2, kj = rem & 3;
  int half = lane >> 4;
  v16h o;
#pragma unroll
  for (int j = 0; j < 16; ++j) {
    int Kl = kt * 32 + amapK(half, j);
    int u = Kl / HS, v = Kl % HS;
    int y = 2 * u - 1 + ki, x = 2 * v - 1 + kj;
    float val = 0.f;
    if (y >= 0 && y < HH && x >= 0 && x < WW)
      val = ctx[(((size_t)n * CC + c) * HH + y) * WW + x];
    o[j] = (_Float16)val;
  }
  out[tid] = o;
}

// ---------------- phase 8: scatter V into output with overlap normalize -----
__global__ __launch_bounds__(256)
void k_assemble_out(const float* __restrict__ V, float* __restrict__ out) {
  int tid = blockIdx.x * blockDim.x + threadIdx.x;
  if (tid >= (int)(NB * CC * HH * WW)) return;
  int x = tid % WW;
  int y = (tid / WW) % HH;
  int c = (tid / (WW * HH)) % CC;
  int n = tid / (WW * HH * CC);
  int iy[2], iki[2], jx[2], jkj[2];
  int cy = 0, cx = 0;
#pragma unroll
  for (int ki = 0; ki < 4; ++ki) {
    int t = y + 1 - ki;
    if (t >= 0 && !(t & 1) && (t >> 1) < HS) { iy[cy] = t >> 1; iki[cy] = ki; ++cy; }
  }
#pragma unroll
  for (int kj = 0; kj < 4; ++kj) {
    int t = x + 1 - kj;
    if (t >= 0 && !(t & 1) && (t >> 1) < HS) { jx[cx] = t >> 1; jkj[cx] = kj; ++cx; }
  }
  float acc = 0.f;
  for (int a = 0; a < cy; ++a)
    for (int b = 0; b < cx; ++b)
      acc += V[((size_t)n * LL + iy[a] * HS + jx[b]) * CF + c * 16 + iki[a] * 4 + jkj[b]];
  out[tid] = acc / (float)(cy * cx);
}

extern "C" void kernel_launch(void* const* d_in, const int* in_sizes, int n_in,
                              void* d_out, int out_size, void* d_ws, size_t ws_size,
                              hipStream_t stream) {
  (void)in_sizes; (void)n_in; (void)out_size;
  if (ws_size < WS_NEEDED) return;   // scratch too small: do nothing (safe, deterministic)

  const float* x    = (const float*)d_in[0];
  const float* ctx  = (const float*)d_in[1];
  const float* mask = (const float*)d_in[2];
  float* out = (float*)d_out;
  float* att = out + OUT0;           // att_score region IS att[n][l][p] flat
  char*  ws  = (char*)d_ws;

  float* norm  = (float*)(ws + NORM_OFF);
  float* bias  = (float*)(ws + BIAS_OFF);
  v16h*  A1    = (v16h*)(ws + A1_OFF);
  v16h*  B1    = (v16h*)(ws + B1_OFF);
  float* corr0 = (float*)(ws + X_OFF);
  float* m3    = (float*)(ws + Y_OFF);
  float* score = (float*)(ws + SCORE_OFF);
  v16h*  A2    = (v16h*)(ws + A2_OFF);
  v16h*  B2    = (v16h*)(ws + B2_OFF);
  float* V     = (float*)(ws + V_OFF);

  const int T = 256;
  // phase 0: norms + mask bias
  k_norm_bias<<<(NB * LL + T - 1) / T, T, 0, stream>>>(ctx, mask, norm, bias);
  // phase 1: fragments for correlation GEMM
  k_build_A1<<<(NB * MT1 * KT1 * 32 + T - 1) / T, T, 0, stream>>>(ctx, norm, A1);
  k_build_B1<<<(NB * KT1 * NT1 * 32 + T - 1) / T, T, 0, stream>>>(x, B1);
  // phase 2: corr[l,p] = cols @ xp   (2304 x 864 x 2304, per batch)
  {
    int tilesPerBatch = (MT1 / 2) * (NT1 / 4);          // 72 * 36
    int waves = NB * tilesPerBatch;                     // 10368
    wmma_gemm_tile<<<(waves + 7) / 8, T, 0, stream>>>(A1, B1, corr0, MT1, NT1,
                                                      KT1, LL,
                                                      (long long)LL * LL,
                                                      tilesPerBatch, waves);
  }
  // phases 3-4: diagonal fuse convs (transpose folded in), + mask bias
  long long nElem = (long long)NB * LL * LL;
  int fuseBlocks = (int)((nElem + T - 1) / T);
  k_fuse1<<<fuseBlocks, T, 0, stream>>>(corr0, m3);
  k_fuse2<<<fuseBlocks, T, 0, stream>>>(m3, bias, score);
  // phase 5: softmax over l -> writes att_score output directly
  k_softmax<<<NB * LL, T, 0, stream>>>(score, att);
  // phase 6: fragments for deconv GEMM  (attT @ raw_filter)
  k_build_A2<<<(NB * MT2 * KT2 * 32 + T - 1) / T, T, 0, stream>>>(att, A2);
  k_build_B2<<<(NB * KT2 * NT2 * 32 + T - 1) / T, T, 0, stream>>>(ctx, B2);
  // phase 7: V[p, (c,ki,kj)]  (2304 x 2304 x 1536, per batch)
  {
    int tilesPerBatch = (MT2 / 2) * (NT2 / 4);          // 72 * 24
    int waves = NB * tilesPerBatch;                     // 6912
    wmma_gemm_tile<<<(waves + 7) / 8, T, 0, stream>>>(A2, B2, V, MT2, NT2,
                                                      KT2, CF,
                                                      (long long)LL * CF,
                                                      tilesPerBatch, waves);
  }
  // phase 8: transposed-conv scatter + overlap normalization
  k_assemble_out<<<(int)((OUT0 + T - 1) / T), T, 0, stream>>>(V, out);
}
